// Block_23055384445882
// MI455X (gfx1250) — compile-verified, matrix operand order checked by reading, not run on hardware
//
#include <hip/hip_runtime.h>

// ---------------------------------------------------------------------------
// Types for CDNA5 WMMA (wave32, 16x16x32 bf16 -> f32) and TDM descriptors
// ---------------------------------------------------------------------------
typedef __bf16 bf16_t;
typedef __attribute__((ext_vector_type(16))) __bf16        v16bf;
typedef __attribute__((ext_vector_type(8)))  float         v8f;
typedef __attribute__((ext_vector_type(4)))  unsigned int  v4u;
typedef __attribute__((ext_vector_type(8)))  int           i32x8;
typedef __attribute__((ext_vector_type(4)))  int           i32x4;

union Frag { v4u u[2]; v16bf b; };

__device__ __forceinline__ v8f wmma_bf16(v16bf a, v16bf b, v8f c) {
  // (neg_a, A, neg_b, B, c_mod, C, reuse_a, reuse_b)
  return __builtin_amdgcn_wmma_f32_16x16x32_bf16(false, a, false, b, (short)0, c, false, false);
}

__device__ __forceinline__ float silu_f(float v) { return v / (1.f + __expf(-v)); }

// ---------------------------------------------------------------------------
// Problem constants (from reference)
// ---------------------------------------------------------------------------
constexpr int Bb   = 2;
constexpr int Tt   = 2048;
constexpr int Cc   = 2048;
constexpr int NH   = 16;
constexpr int NG   = 4;
constexpr int HS   = 128;
constexpr int IM   = 5632;
constexpr int MR   = Bb * Tt;              // 4096 rows
constexpr int QKVN = (NH + 2 * NG) * HS;   // 3072

// ---------------------------------------------------------------------------
// TDM: issue one 2D tile load (tileH rows x tileW bf16 elements) into LDS.
// D# bitfields per CDNA5 ISA ch.8: group0 {count=1, lds_addr, global_addr,
// type=2}; group1 {data_size=2B, tensor dims, tile dims, row stride}.
// All operands are wave-uniform; EXEC is ignored by TDM.
// This toolchain uses the 6-arg builtin: (g0, g1, g2, g3, g_extra, cpol).
// ---------------------------------------------------------------------------
__device__ __forceinline__ void tdm_load_2d(unsigned ldsOff, const void* gptr,
                                            unsigned rows, unsigned rowLen,
                                            unsigned long long strideElems,
                                            unsigned tileW, unsigned tileH) {
  const unsigned long long ga = (unsigned long long)gptr;
  v4u g0;
  g0.x = 1u;                                                   // count=1 (valid)
  g0.y = ldsOff;                                               // lds_addr (bytes)
  g0.z = (unsigned)ga;                                         // global_addr[31:0]
  g0.w = ((unsigned)(ga >> 32) & 0x01FFFFFFu) | 0x80000000u;   // ga[56:32] | type=2
  i32x8 g1;
  g1[0] = (int)(1u << 16);                                     // data_size = 2 bytes
  g1[1] = (int)((rowLen & 0xFFFFu) << 16);                     // tensor_dim0[15:0]
  g1[2] = (int)((rowLen >> 16) | ((rows & 0xFFFFu) << 16));    // td0[31:16] | td1[15:0]
  g1[3] = (int)((rows >> 16) | (tileW << 16));                 // td1[31:16] | tile_dim0
  g1[4] = (int)tileH;                                          // tile_dim1 (tile_dim2=0)
  g1[5] = (int)(unsigned)strideElems;                          // tensor_dim0_stride[31:0]
  g1[6] = (int)((unsigned)(strideElems >> 32) & 0xFFFFu);      // stride[47:32]
  g1[7] = 0;
  i32x4 z4 = {0, 0, 0, 0};
  i32x8 z8 = {0, 0, 0, 0, 0, 0, 0, 0};
  __builtin_amdgcn_tensor_load_to_lds(g0, g1, z4, z4, z8, 0);
}

__device__ __forceinline__ unsigned lds_off(const void* p) {
  // Flat LDS aperture keeps the byte offset in addr[31:0] (ISA 10.2)
  return (unsigned)(unsigned long long)p;
}

// ---------------------------------------------------------------------------
// fp32 -> bf16 cast (weights, once per launch)
// ---------------------------------------------------------------------------
__global__ void cast_f32_bf16(const float* __restrict__ in, bf16_t* __restrict__ out, int n) {
  int i = blockIdx.x * blockDim.x + threadIdx.x;
  int stride = gridDim.x * blockDim.x;
  for (; i < n; i += stride) out[i] = (bf16_t)in[i];
}

// ---------------------------------------------------------------------------
// RMSNorm (fp32 in, bf16 out), one block per row
// ---------------------------------------------------------------------------
__global__ __launch_bounds__(256)
void rmsnorm_bf16(const float* __restrict__ x, const float* __restrict__ w,
                  bf16_t* __restrict__ out, int ncols) {
  __shared__ float sred[8];
  __shared__ float sscale;
  const int row = blockIdx.x;
  const float* xr = x + (size_t)row * ncols;
  float ss = 0.f;
  for (int i = threadIdx.x; i < ncols; i += blockDim.x) { float v = xr[i]; ss = fmaf(v, v, ss); }
  for (int m = 16; m >= 1; m >>= 1) ss += __shfl_xor(ss, m, 32);
  const int lane = threadIdx.x & 31, wv = threadIdx.x >> 5;
  if (lane == 0) sred[wv] = ss;
  __syncthreads();
  if (threadIdx.x == 0) {
    float t = 0.f;
    for (int i = 0; i < (int)(blockDim.x >> 5); ++i) t += sred[i];
    sscale = rsqrtf(t / (float)ncols + 1e-5f);
  }
  __syncthreads();
  const float sc = sscale;
  bf16_t* orow = out + (size_t)row * ncols;
  for (int i = threadIdx.x; i < ncols; i += blockDim.x) orow[i] = (bf16_t)(xr[i] * sc * w[i]);
}

// ---------------------------------------------------------------------------
// TDM-staged WMMA GEMM: D[M,N] = A[M,K] @ B[N,K]^T  (A,B bf16, K-major rows)
// Block 256 threads (8 waves, 4Mx2N), block tile 128x128, K-step 32.
// TDM stages A(128x32) / B0(128x32) / (B1) tiles into double-buffered LDS;
// waves 0..NB-1 each own one descriptor per stage (TENSORcnt is per-wave,
// completes in order). Wave tile 32x64 -> 8 WMMAs per stage (16 for MODE 2).
// MODE 0: store fp32 | MODE 1: fp32 + residual | MODE 2: bf16 silu(d0)*d1
// ---------------------------------------------------------------------------
template <int MODE>
__global__ __launch_bounds__(256)
void gemm_wmma(const bf16_t* __restrict__ A, const bf16_t* __restrict__ B0,
               const bf16_t* __restrict__ B1, const float* __restrict__ resid,
               void* __restrict__ outp, int Mdim, int Ndim, int Kdim) {
  constexpr int NB = (MODE == 2) ? 3 : 2;    // tiles per stage: A, B0, (B1)
  constexpr int TILE = 128 * 32;             // elements per staged tile
  __shared__ __align__(16) bf16_t sm[2][NB][TILE];

  const int lane = threadIdx.x & 31;
  const int wave = threadIdx.x >> 5;
  const int wm = wave & 3;                   // 4 waves over M
  const int wn = wave >> 2;                  // 2 waves over N
  const int row0 = blockIdx.y * 128;
  const int col0 = blockIdx.x * 128;
  const int hi = lane >> 4, l16 = lane & 15;

  const int S = Kdim >> 5;                   // K stages of 32

  v8f acc[2][4];
  v8f acc2[2][4];
#pragma unroll
  for (int rb = 0; rb < 2; ++rb)
#pragma unroll
    for (int cb = 0; cb < 4; ++cb)
#pragma unroll
      for (int r = 0; r < 8; ++r) { acc[rb][cb][r] = 0.f; if (MODE == 2) acc2[rb][cb][r] = 0.f; }

  // Per-wave TDM issue: wave0 -> A tile, wave1 -> B0 tile, wave2 -> B1 tile
  auto issue = [&](int s) {
    const int st = s & 1;
    const int k0 = s << 5;
    if (wave == 0) {
      tdm_load_2d(lds_off(&sm[st][0][0]), A + (size_t)row0 * Kdim + k0,
                  (unsigned)(Mdim - row0), (unsigned)(Kdim - k0),
                  (unsigned long long)Kdim, 32u, 128u);
    } else if (wave == 1) {
      tdm_load_2d(lds_off(&sm[st][1][0]), B0 + (size_t)col0 * Kdim + k0,
                  (unsigned)(Ndim - col0), (unsigned)(Kdim - k0),
                  (unsigned long long)Kdim, 32u, 128u);
    } else if (MODE == 2 && wave == 2) {
      tdm_load_2d(lds_off(&sm[st][2][0]), B1 + (size_t)col0 * Kdim + k0,
                  (unsigned)(Ndim - col0), (unsigned)(Kdim - k0),
                  (unsigned long long)Kdim, 32u, 128u);
    }
  };

  issue(0);
  for (int s = 0; s < S; ++s) {
    if (s + 1 < S) {
      issue(s + 1);                              // prefetch next stage (other buffer)
      if (wave < NB) __builtin_amdgcn_s_wait_tensorcnt((short)1);  // stage s done
    } else {
      if (wave < NB) __builtin_amdgcn_s_wait_tensorcnt((short)0);
    }
    __syncthreads();                             // staged tiles visible to all waves

    const bf16_t* Abuf  = &sm[s & 1][0][0];
    const bf16_t* Bbuf  = &sm[s & 1][1][0];
    const bf16_t* B1buf = (MODE == 2) ? &sm[s & 1][NB - 1][0] : nullptr;

    Frag af[2];
#pragma unroll
    for (int rb = 0; rb < 2; ++rb) {
      const bf16_t* ap = Abuf + (wm * 32 + rb * 16 + l16) * 32 + hi * 8;
      af[rb].u[0] = *(const v4u*)(ap);
      af[rb].u[1] = *(const v4u*)(ap + 16);
    }
#pragma unroll
    for (int cb = 0; cb < 4; ++cb) {
      const bf16_t* bp = Bbuf + (wn * 64 + cb * 16 + l16) * 32 + hi * 8;
      Frag bf0;
      bf0.u[0] = *(const v4u*)(bp);
      bf0.u[1] = *(const v4u*)(bp + 16);
      acc[0][cb] = wmma_bf16(af[0].b, bf0.b, acc[0][cb]);
      acc[1][cb] = wmma_bf16(af[1].b, bf0.b, acc[1][cb]);
      if (MODE == 2) {
        const bf16_t* bp1 = B1buf + (wn * 64 + cb * 16 + l16) * 32 + hi * 8;
        Frag bf1;
        bf1.u[0] = *(const v4u*)(bp1);
        bf1.u[1] = *(const v4u*)(bp1 + 16);
        acc2[0][cb] = wmma_bf16(af[0].b, bf1.b, acc2[0][cb]);
        acc2[1][cb] = wmma_bf16(af[1].b, bf1.b, acc2[1][cb]);
      }
    }
    __syncthreads();                             // done reading before overwrite
  }

  // Epilogue. D layout: lane l16 = column, rows r + 8*hi
#pragma unroll
  for (int rb = 0; rb < 2; ++rb) {
    const int rBase = row0 + wm * 32 + rb * 16 + hi * 8;
#pragma unroll
    for (int cb = 0; cb < 4; ++cb) {
      const int col = col0 + wn * 64 + cb * 16 + l16;
#pragma unroll
      for (int r = 0; r < 8; ++r) {
        const size_t idx = (size_t)(rBase + r) * Ndim + col;
        if (MODE == 0) {
          ((float*)outp)[idx] = acc[rb][cb][r];
        } else if (MODE == 1) {
          ((float*)outp)[idx] = resid[idx] + acc[rb][cb][r];
        } else {
          ((bf16_t*)outp)[idx] = (bf16_t)(silu_f(acc[rb][cb][r]) * acc2[rb][cb][r]);
        }
      }
    }
  }
}

// ---------------------------------------------------------------------------
// Depthwise causal conv(4) + SiLU + RoPE + repack:
//   q -> Qb [B,NH,T,HS] bf16 (roped)
//   k -> Kb [B,NG,T,HS] bf16 (roped)
//   v -> Vt [B,NG,HS,T] bf16 (transposed, no rope)
// ---------------------------------------------------------------------------
__global__ __launch_bounds__(64)
void conv_rope_pack(const float* __restrict__ qkv,
                    const float* __restrict__ qw, const float* __restrict__ kw,
                    const float* __restrict__ vw,
                    const float* __restrict__ cosw, const float* __restrict__ sinw,
                    bf16_t* __restrict__ Qb, bf16_t* __restrict__ Kb, bf16_t* __restrict__ Vt) {
  constexpr int TT = 32;
  constexpr int tTiles = Tt / TT;  // 64
  int idx = blockIdx.x;
  const int tt   = idx % tTiles; idx /= tTiles;
  const int slot = idx % 6;      idx /= 6;
  const int g    = idx % NG;     idx /= NG;
  const int b    = idx;
  const int d    = threadIdx.x;  // 0..63

  const int ch0 = (g * 6 + slot) * HS + d;
  const int ch1 = ch0 + 64;

  const float* cw; int cb;
  if (slot < 4)       { cw = qw; cb = (g * 4 + slot) * HS; }
  else if (slot == 4) { cw = kw; cb = g * HS; }
  else                { cw = vw; cb = g * HS; }

  float w0[4], w1[4];
#pragma unroll
  for (int j = 0; j < 4; ++j) {
    w0[j] = cw[(cb + d) * 4 + j];
    w1[j] = cw[(cb + d + 64) * 4 + j];
  }

  const int t0 = tt * TT;
  auto xv = [&](int t, int ch) -> float {
    return (t < 0) ? 0.f : qkv[((size_t)(b * Tt + t)) * QKVN + ch];
  };

  float a0 = xv(t0 - 3, ch0), b0v = xv(t0 - 2, ch0), c0 = xv(t0 - 1, ch0);
  float a1 = xv(t0 - 3, ch1), b1v = xv(t0 - 2, ch1), c1 = xv(t0 - 1, ch1);

  for (int t = t0; t < t0 + TT; ++t) {
    const float d0 = xv(t, ch0), d1 = xv(t, ch1);
    float y0 = w0[0] * a0 + w0[1] * b0v + w0[2] * c0 + w0[3] * d0;
    float y1 = w1[0] * a1 + w1[1] * b1v + w1[2] * c1 + w1[3] * d1;
    y0 = silu_f(y0); y1 = silu_f(y1);

    if (slot == 5) {
      const size_t base = (size_t)(b * NG + g) * HS;
      Vt[(base + d) * Tt + t]      = (bf16_t)y0;
      Vt[(base + d + 64) * Tt + t] = (bf16_t)y1;
    } else {
      const float cc = cosw[t * (HS / 2) + d];
      const float ssn = sinw[t * (HS / 2) + d];
      const float o1 = y0 * cc - y1 * ssn;
      const float o2 = y0 * ssn + y1 * cc;
      if (slot < 4) {
        const int h = g * 4 + slot;
        const size_t base = ((size_t)((b * NH + h) * Tt) + t) * HS;
        Qb[base + d] = (bf16_t)o1;
        Qb[base + d + 64] = (bf16_t)o2;
      } else {
        const size_t base = ((size_t)((b * NG + g) * Tt) + t) * HS;
        Kb[base + d] = (bf16_t)o1;
        Kb[base + d + 64] = (bf16_t)o2;
      }
    }
    a0 = b0v; b0v = c0; c0 = d0;
    a1 = b1v; b1v = c1; c1 = d1;
  }
}

// ---------------------------------------------------------------------------
// Flash attention (causal, GQA). One wave per (b, h, 16-row query tile).
// Key tiles of 32. S via WMMA, P routed through LDS into the A-operand
// layout, O += P*V via WMMA (V stored transposed: [B,NG,HS,T]).
// ---------------------------------------------------------------------------
__global__ __launch_bounds__(128)
void attn_wmma(const bf16_t* __restrict__ Qb, const bf16_t* __restrict__ Kb,
               const bf16_t* __restrict__ Vt, bf16_t* __restrict__ Y) {
  __shared__ __align__(16) bf16_t plds[4][16][40];  // 40 = 32 + pad (16B-aligned rows)

  const int lane = threadIdx.x & 31;
  const int wave = threadIdx.x >> 5;
  const int gid  = blockIdx.x * 4 + wave;
  constexpr int qTiles = Tt / 16;  // 128
  const int qt = gid % qTiles;
  const int h  = (gid / qTiles) % NH;
  const int b  = gid / (qTiles * NH);
  const int g  = h / (NH / NG);
  const int q0 = qt * 16;
  const int hi  = lane >> 4;
  const int l16 = lane & 15;

  const bf16_t* qrow = Qb + ((size_t)((b * NH + h) * Tt) + q0 + l16) * HS + hi * 8;
  Frag qf[4];
#pragma unroll
  for (int c = 0; c < 4; ++c) {
    qf[c].u[0] = *(const v4u*)(qrow + c * 32);
    qf[c].u[1] = *(const v4u*)(qrow + c * 32 + 16);
  }

  v8f o[8];
  float rmax[8], rsum[8];
#pragma unroll
  for (int c = 0; c < 8; ++c)
#pragma unroll
    for (int r = 0; r < 8; ++r) o[c][r] = 0.f;
#pragma unroll
  for (int r = 0; r < 8; ++r) { rmax[r] = -3.0e38f; rsum[r] = 0.f; }

  const bf16_t* kbase = Kb + (size_t)(b * NG + g) * Tt * HS;
  const bf16_t* vbase = Vt + (size_t)(b * NG + g) * HS * Tt;
  const float scale = 0.08838834764831845f;  // 1/sqrt(128)

  const int nkt = (q0 + 16 + 31) / 32;  // causal: key tiles needed
  for (int kt = 0; kt < nkt; ++kt) {
    const int kBase = kt * 32;
    v8f s0, s1;
#pragma unroll
    for (int r = 0; r < 8; ++r) { s0[r] = 0.f; s1[r] = 0.f; }

    const bf16_t* krow0 = kbase + (size_t)(kBase + l16) * HS + hi * 8;
    const bf16_t* krow1 = kbase + (size_t)(kBase + 16 + l16) * HS + hi * 8;
#pragma unroll
    for (int c = 0; c < 4; ++c) {
      Frag kf0, kf1;
      kf0.u[0] = *(const v4u*)(krow0 + c * 32);
      kf0.u[1] = *(const v4u*)(krow0 + c * 32 + 16);
      kf1.u[0] = *(const v4u*)(krow1 + c * 32);
      kf1.u[1] = *(const v4u*)(krow1 + c * 32 + 16);
      s0 = wmma_bf16(qf[c].b, kf0.b, s0);
      s1 = wmma_bf16(qf[c].b, kf1.b, s1);
    }

    const bool maskTile = (kBase + 31 > q0 + 15);
    const int rowg = q0 + hi * 8;  // + r
    float p0[8], p1[8], nmax[8];
#pragma unroll
    for (int r = 0; r < 8; ++r) {
      float v0 = s0[r] * scale;
      float v1 = s1[r] * scale;
      if (maskTile) {
        if (kBase + l16 > rowg + r)       v0 = -3.0e38f;
        if (kBase + 16 + l16 > rowg + r)  v1 = -3.0e38f;
      }
      float mx = fmaxf(v0, v1);
      for (int m = 8; m >= 1; m >>= 1) mx = fmaxf(mx, __shfl_xor(mx, m, 32));
      nmax[r] = fmaxf(rmax[r], mx);
      p0[r] = __expf(v0 - nmax[r]);
      p1[r] = __expf(v1 - nmax[r]);
    }

#pragma unroll
    for (int r = 0; r < 8; ++r) {
      const float cf = __expf(rmax[r] - nmax[r]);
      float ps = p0[r] + p1[r];
      for (int m = 8; m >= 1; m >>= 1) ps += __shfl_xor(ps, m, 32);
      rsum[r] = rsum[r] * cf + ps;
      rmax[r] = nmax[r];
#pragma unroll
      for (int c = 0; c < 8; ++c) o[c][r] *= cf;
      plds[wave][hi * 8 + r][l16]      = (bf16_t)p0[r];
      plds[wave][hi * 8 + r][l16 + 16] = (bf16_t)p1[r];
    }

    Frag pf;
    pf.u[0] = *(const v4u*)&plds[wave][l16][hi * 8];
    pf.u[1] = *(const v4u*)&plds[wave][l16][hi * 8 + 16];

#pragma unroll
    for (int c = 0; c < 8; ++c) {
      const bf16_t* vrow = vbase + (size_t)(c * 16 + l16) * Tt + kBase + hi * 8;
      Frag vf;
      vf.u[0] = *(const v4u*)(vrow);
      vf.u[1] = *(const v4u*)(vrow + 16);
      o[c] = wmma_bf16(pf.b, vf.b, o[c]);
    }
  }

#pragma unroll
  for (int c = 0; c < 8; ++c) {
#pragma unroll
    for (int r = 0; r < 8; ++r) {
      const int rowg = q0 + hi * 8 + r;
      const float val = o[c][r] / rsum[r];
      Y[((size_t)(b * Tt) + rowg) * Cc + h * HS + c * 16 + l16] = (bf16_t)val;
    }
  }
}

// ---------------------------------------------------------------------------
// Host: workspace carving + launches
// ---------------------------------------------------------------------------
static inline char* ws_take(char*& p, size_t bytes) {
  char* r = p;
  p += (bytes + 255) & ~(size_t)255;
  return r;
}

extern "C" void kernel_launch(void* const* d_in, const int* in_sizes, int n_in,
                              void* d_out, int out_size, void* d_ws, size_t ws_size,
                              hipStream_t stream) {
  const float* x      = (const float*)d_in[0];
  const float* cosw   = (const float*)d_in[1];
  const float* sinw   = (const float*)d_in[2];
  const float* n1w    = (const float*)d_in[3];
  const float* attn_w = (const float*)d_in[4];
  const float* proj_w = (const float*)d_in[5];
  const float* qconv  = (const float*)d_in[6];
  const float* kconv  = (const float*)d_in[7];
  const float* vconv  = (const float*)d_in[8];
  const float* n2w    = (const float*)d_in[9];
  const float* fc1_w  = (const float*)d_in[10];
  const float* fc2_w  = (const float*)d_in[11];
  const float* mlpp_w = (const float*)d_in[12];

  char* p = (char*)d_ws;
  bf16_t* wAttn = (bf16_t*)ws_take(p, (size_t)QKVN * Cc * 2);      // 3072x2048
  bf16_t* wProj = (bf16_t*)ws_take(p, (size_t)Cc * Cc * 2);        // 2048x2048
  bf16_t* wFc1  = (bf16_t*)ws_take(p, (size_t)IM * Cc * 2);        // 5632x2048
  bf16_t* wFc2  = (bf16_t*)ws_take(p, (size_t)IM * Cc * 2);
  bf16_t* wMlp  = (bf16_t*)ws_take(p, (size_t)Cc * IM * 2);        // 2048x5632
  bf16_t* n12   = (bf16_t*)ws_take(p, (size_t)MR * Cc * 2);        // n1, reused as n2
  float*  qkv   = (float*) ws_take(p, (size_t)MR * QKVN * 4);      // reused as gate bf16
  bf16_t* Qb    = (bf16_t*)ws_take(p, (size_t)Bb * NH * Tt * HS * 2);
  bf16_t* Kbuf  = (bf16_t*)ws_take(p, (size_t)Bb * NG * Tt * HS * 2);
  bf16_t* Vt    = (bf16_t*)ws_take(p, (size_t)Bb * NG * HS * Tt * 2);
  bf16_t* Yb    = (bf16_t*)ws_take(p, (size_t)MR * Cc * 2);
  float*  X1    = (float*) ws_take(p, (size_t)MR * Cc * 4);
  bf16_t* gate  = (bf16_t*)qkv;  // 4096x5632 bf16 (46.1MB) fits in qkv f32 (50.3MB)

  // 1) weights -> bf16
  cast_f32_bf16<<<2048, 256, 0, stream>>>(attn_w, wAttn, QKVN * Cc);
  cast_f32_bf16<<<2048, 256, 0, stream>>>(proj_w, wProj, Cc * Cc);
  cast_f32_bf16<<<2048, 256, 0, stream>>>(fc1_w,  wFc1,  IM * Cc);
  cast_f32_bf16<<<2048, 256, 0, stream>>>(fc2_w,  wFc2,  IM * Cc);
  cast_f32_bf16<<<2048, 256, 0, stream>>>(mlpp_w, wMlp,  Cc * IM);

  // 2) n1 = rmsnorm(x)
  rmsnorm_bf16<<<MR, 256, 0, stream>>>(x, n1w, n12, Cc);

  // 3) qkv = n1 @ attn_w^T   [4096 x 3072]
  gemm_wmma<0><<<dim3(QKVN / 128, MR / 128), 256, 0, stream>>>(
      n12, wAttn, nullptr, nullptr, (void*)qkv, MR, QKVN, Cc);

  // 4) conv + silu + rope + repack
  conv_rope_pack<<<Bb * NG * 6 * (Tt / 32), 64, 0, stream>>>(
      qkv, qconv, kconv, vconv, cosw, sinw, Qb, Kbuf, Vt);

  // 5) flash attention -> Y bf16
  attn_wmma<<<(Bb * NH * (Tt / 16)) / 4, 128, 0, stream>>>(Qb, Kbuf, Vt, Yb);

  // 6) x1 = x + Y @ proj_w^T
  gemm_wmma<1><<<dim3(Cc / 128, MR / 128), 256, 0, stream>>>(
      Yb, wProj, nullptr, x, (void*)X1, MR, Cc, Cc);

  // 7) n2 = rmsnorm(x1)
  rmsnorm_bf16<<<MR, 256, 0, stream>>>(X1, n2w, n12, Cc);

  // 8) gate = silu(n2 @ fc1^T) * (n2 @ fc2^T)   [4096 x 5632] bf16
  gemm_wmma<2><<<dim3(IM / 128, MR / 128), 256, 0, stream>>>(
      n12, wFc1, wFc2, nullptr, (void*)gate, MR, IM, Cc);

  // 9) out = x1 + gate @ mlp_proj^T
  gemm_wmma<1><<<dim3(Cc / 128, MR / 128), 256, 0, stream>>>(
      gate, wMlp, nullptr, X1, d_out, MR, Cc, IM);
}
